// MultiheadLinformer_86139864089110
// MI455X (gfx1250) — matrix operand order repaired and checked by reading, not compiled, not run on hardware
//
#include <hip/hip_runtime.h>
#include <hip/hip_bf16.h>

typedef __bf16 bf16_t;
typedef __attribute__((ext_vector_type(16))) __bf16 v16bf;
typedef __attribute__((ext_vector_type(8)))  __bf16 v8bf;
typedef __attribute__((ext_vector_type(8)))  float  v8f;
typedef __attribute__((ext_vector_type(4)))  float  v4f;

#define HH 16
#define DD 1024
#define EE 64
#define NN 4096
#define KK 256

// ---------------------------------------------------------------------------
// WMMA helpers (bf16 in, f32 accumulate), wave32, 16x16x32 shape.
// A-fragment: lane L holds row M=L&15; elements 0..15 = K (L>>4)*16 .. +15.
// B-fragment: lane L holds col N=L&15; elements 0..15 = K (L>>4)*16 .. +15,
//             sourced from a transposed (N-major) array so loads are b128.
// C-fragment: element r at (M = r + 8*(L>>4), N = L&15).
// ---------------------------------------------------------------------------

static __device__ __forceinline__ v8f wmma_bf16(v16bf a, v16bf b, v8f c) {
  return __builtin_amdgcn_wmma_f32_16x16x32_bf16(
      /*neg_a=*/false, a, /*neg_b=*/false, b,
      /*c_mod=*/(short)0, c, /*reuse_a=*/false, /*reuse_b=*/false);
}

static __device__ __forceinline__ v16bf load_a_frag(const bf16_t* __restrict__ tile,
                                                    int lda, int lane) {
  const bf16_t* p = tile + (size_t)(lane & 15) * lda + ((lane >> 4) << 4);
  v8bf lo = *(const v8bf*)(p);
  v8bf hi = *(const v8bf*)(p + 8);
  v16bf r;
#pragma unroll
  for (int i = 0; i < 8; ++i) { r[i] = lo[i]; r[i + 8] = hi[i]; }
  return r;
}

// BT is stored N-major: BT[n][k] == B[k][n]; tile points at (n0, k0).
static __device__ __forceinline__ v16bf load_b_frag(const bf16_t* __restrict__ tile,
                                                    int ldb, int lane) {
  const bf16_t* p = tile + (size_t)(lane & 15) * ldb + ((lane >> 4) << 4);
  v8bf lo = *(const v8bf*)(p);
  v8bf hi = *(const v8bf*)(p + 8);
  v16bf r;
#pragma unroll
  for (int i = 0; i < 8; ++i) { r[i] = lo[i]; r[i + 8] = hi[i]; }
  return r;
}

// Async global -> LDS copy of one 16-byte chunk (VGLOBAL op, tracked by ASYNCcnt).
static __device__ __forceinline__ void async_copy_b128(unsigned lds_off,
                                                       const void* gaddr) {
  asm volatile("global_load_async_to_lds_b128 %0, %1, off"
               :: "v"(lds_off), "v"(gaddr) : "memory");
}
static __device__ __forceinline__ void wait_asynccnt0() {
  asm volatile("s_wait_asynccnt 0x0" ::: "memory");
}

// ---------------------------------------------------------------------------
// Kernels
// ---------------------------------------------------------------------------

// Vectorized fp32 -> bf16 cast: 8 elements/thread/iter (all sizes are 8-multiples).
__global__ void cast_f32_bf16_kernel(const float* __restrict__ src,
                                     bf16_t* __restrict__ dst, size_t n) {
  size_t i = ((size_t)blockIdx.x * blockDim.x + threadIdx.x) * 8;
  size_t stride = (size_t)gridDim.x * blockDim.x * 8;
  for (; i + 8 <= n; i += stride) {
    v4f lo = *(const v4f*)(src + i);
    v4f hi = *(const v4f*)(src + i + 4);
    v8bf o;
#pragma unroll
    for (int j = 0; j < 4; ++j) { o[j] = (bf16_t)lo[j]; o[j + 4] = (bf16_t)hi[j]; }
    *(v8bf*)(dst + i) = o;
  }
}

// q = x @ Wq^T + bq (row-major (N,E) per head, bf16)
// k,v likewise but stored TRANSPOSED (E,N) per head for the kp/vp GEMMs.
// One wave computes a 32x64 macro-tile: 2 M-frags x 4 N-frags, 8 WMMA / K-step.
__global__ void qkv_kernel(const bf16_t* __restrict__ xb,
                           const bf16_t* __restrict__ Wqb,
                           const bf16_t* __restrict__ Wkb,
                           const bf16_t* __restrict__ Wvb,
                           const float* __restrict__ bq,
                           const float* __restrict__ bk,
                           const float* __restrict__ bv,
                           bf16_t* __restrict__ qb,
                           bf16_t* __restrict__ kTb,
                           bf16_t* __restrict__ vTb) {
  int lane = threadIdx.x & 31;
  int w = blockIdx.x * (blockDim.x >> 5) + (threadIdx.x >> 5);
  int mt2 = w & 127; w >>= 7;   // 32-row strip (128 of them)
  int h   = w & 15;  w >>= 4;   // head
  int t   = w;                  // 0=q 1=k 2=v
  if (t > 2) return;

  const bf16_t* W    = (t == 0) ? Wqb : (t == 1) ? Wkb : Wvb;
  const float*  bias = (t == 0) ? bq  : (t == 1) ? bk  : bv;
  const bf16_t* A0 = xb + (size_t)(mt2 * 32) * DD;
  const bf16_t* A1 = A0 + (size_t)16 * DD;
  const bf16_t* B0 = W + (size_t)h * EE * DD;   // (E,D) row-major == BT

  v8f acc[2][4] = {};
  for (int kk = 0; kk < DD; kk += 32) {
    __builtin_prefetch(A0 + kk + 256, 0, 3);
    __builtin_prefetch(B0 + kk + 256, 0, 3);
    v16bf a0 = load_a_frag(A0 + kk, DD, lane);
    v16bf a1 = load_a_frag(A1 + kk, DD, lane);
#pragma unroll
    for (int e = 0; e < 4; ++e) {
      v16bf b = load_b_frag(B0 + (size_t)(e * 16) * DD + kk, DD, lane);
      acc[0][e] = wmma_bf16(a0, b, acc[0][e]);
      acc[1][e] = wmma_bf16(a1, b, acc[1][e]);
    }
  }

#pragma unroll
  for (int mi = 0; mi < 2; ++mi) {
    int n0 = mt2 * 32 + mi * 16;
#pragma unroll
    for (int e = 0; e < 4; ++e) {
      float be = bias[h * EE + e * 16 + (lane & 15)];
      if (t == 0) {
        bf16_t* out = qb + (size_t)h * NN * EE;
#pragma unroll
        for (int r = 0; r < 8; ++r) {
          int m = n0 + r + ((lane >> 4) << 3);
          out[(size_t)m * EE + e * 16 + (lane & 15)] = (bf16_t)(acc[mi][e][r] + be);
        }
      } else {
        bf16_t* outT = ((t == 1) ? kTb : vTb) + (size_t)h * EE * NN;
        v8bf pack;
#pragma unroll
        for (int r = 0; r < 8; ++r) pack[r] = (bf16_t)(acc[mi][e][r] + be);
        int ee = e * 16 + (lane & 15);
        int m  = n0 + ((lane >> 4) << 3);
        *(v8bf*)(outT + (size_t)ee * NN + m) = pack;  // one b128 store
      }
    }
  }
}

// kp = Ew @ k + Eb  -> (K,E) row-major bf16 (BT for score GEMM)
// vp = Fw @ v + Fb  -> (E,K) transposed bf16 (BT for z GEMM)
// One wave computes a 16x64 macro-tile (1 A + 4 B loads per 4 WMMA).
__global__ void kpvp_kernel(const bf16_t* __restrict__ Ewb,
                            const bf16_t* __restrict__ Fwb,
                            const bf16_t* __restrict__ kTb,
                            const bf16_t* __restrict__ vTb,
                            const float* __restrict__ Eb,
                            const float* __restrict__ Fb,
                            bf16_t* __restrict__ kpb,
                            bf16_t* __restrict__ vpTb) {
  int lane = threadIdx.x & 31;
  int w = blockIdx.x * (blockDim.x >> 5) + (threadIdx.x >> 5);
  int jt = w & 15; w >>= 4;    // K tile (16)
  int h  = w & 15; w >>= 4;    // head
  int t  = w;                  // 0=kp 1=vp
  if (t > 1) return;

  const bf16_t* A0 = ((t == 0) ? Ewb : Fwb) + (size_t)h * KK * NN + (size_t)(jt * 16) * NN;
  const bf16_t* B0 = ((t == 0) ? kTb : vTb) + (size_t)h * EE * NN;

  v8f acc[4] = {};
  for (int kk = 0; kk < NN; kk += 32) {
    __builtin_prefetch(A0 + kk + 256, 0, 3);
    __builtin_prefetch(B0 + kk + 256, 0, 3);
    v16bf a = load_a_frag(A0 + kk, NN, lane);
#pragma unroll
    for (int e = 0; e < 4; ++e) {
      v16bf b = load_b_frag(B0 + (size_t)(e * 16) * NN + kk, NN, lane);
      acc[e] = wmma_bf16(a, b, acc[e]);
    }
  }

  const float* bias = (t == 0) ? Eb : Fb;
#pragma unroll
  for (int e = 0; e < 4; ++e) {
    if (t == 0) {
      bf16_t* out = kpb + (size_t)h * KK * EE;
#pragma unroll
      for (int r = 0; r < 8; ++r) {
        int j = jt * 16 + r + ((lane >> 4) << 3);
        out[(size_t)j * EE + e * 16 + (lane & 15)] = (bf16_t)(acc[e][r] + bias[h * KK + j]);
      }
    } else {
      bf16_t* outT = vpTb + (size_t)h * EE * KK;
      v8bf pack;
#pragma unroll
      for (int r = 0; r < 8; ++r) {
        int j = jt * 16 + r + ((lane >> 4) << 3);
        pack[r] = (bf16_t)(acc[e][r] + bias[h * KK + j]);
      }
      int ee = e * 16 + (lane & 15);
      int j0 = jt * 16 + ((lane >> 4) << 3);
      *(v8bf*)(outT + (size_t)ee * KK + j0) = pack;
    }
  }
}

// Fused: score = q @ kp^T / 8 -> softmax over K=256 -> z = attn @ vp.
// One block (8 waves) per (head, 16-row strip of N). Score strip lives in LDS.
// The vp tile (32 KB) is copied global->LDS with ASYNC loads issued up front,
// overlapping the score GEMM + softmax; the z-GEMM then runs entirely from LDS.
__global__ void attn_kernel(const bf16_t* __restrict__ qb,
                            const bf16_t* __restrict__ kpb,
                            const bf16_t* __restrict__ vpTb,
                            bf16_t* __restrict__ zb) {
  __shared__ alignas(16) float  sc[16 * 256];    // score / exp, f32 (16 KB)
  __shared__ alignas(16) bf16_t at[16 * 256];    // attn weights, bf16 (8 KB)
  __shared__ alignas(16) bf16_t vpl[EE * KK];    // vp tile (E,K) bf16 (32 KB)
  __shared__ float red[16 * 16];

  int tid = threadIdx.x;
  int lane = tid & 31;
  int wave = tid >> 5;
  int mt = blockIdx.x & 255;
  int h  = blockIdx.x >> 8;

  const bf16_t* Q0  = qb   + (size_t)h * NN * EE + (size_t)(mt * 16) * EE;  // (16 x 64)
  const bf16_t* KP  = kpb  + (size_t)h * KK * EE;                           // (K,E) == BT
  const bf16_t* VP0 = vpTb + (size_t)h * EE * KK;                           // (E,K) == BT

  // --- kick off async vp copy: 2048 x b128 chunks, 8 per thread ---
  {
    unsigned lds_base = (unsigned)(size_t)(void*)vpl;  // flat[31:0] == LDS offset
#pragma unroll
    for (int i = 0; i < 8; ++i) {
      int chunk = tid + i * 256;                       // 16-byte chunk index
      async_copy_b128(lds_base + chunk * 16, VP0 + chunk * 8);
    }
  }

  // q A-fragments are invariant across j-tiles: load once.
  v16bf qa0 = load_a_frag(Q0, EE, lane);
  v16bf qa1 = load_a_frag(Q0 + 32, EE, lane);

  // --- score: waves cover the 16 j-tiles (2 each) ---
  for (int jt = wave; jt < 16; jt += 8) {
    v8f acc = {};
    v16bf b0 = load_b_frag(KP + (size_t)(jt * 16) * EE, EE, lane);
    v16bf b1 = load_b_frag(KP + (size_t)(jt * 16) * EE + 32, EE, lane);
    acc = wmma_bf16(qa0, b0, acc);
    acc = wmma_bf16(qa1, b1, acc);
#pragma unroll
    for (int r = 0; r < 8; ++r) {
      int m = r + ((lane >> 4) << 3);
      sc[m * 256 + jt * 16 + (lane & 15)] = acc[r] * 0.125f;  // 1/sqrt(E)
    }
  }
  __syncthreads();

  // --- softmax over 256 cols per row: 16 threads per row, 16 cols each ---
  int row = tid >> 4, sub = tid & 15;
  float m = -1e30f;
#pragma unroll
  for (int t = 0; t < 16; ++t) m = fmaxf(m, sc[row * 256 + sub + 16 * t]);
  red[row * 16 + sub] = m;
  __syncthreads();
  if (sub == 0) {
    float mm = red[row * 16];
#pragma unroll
    for (int i = 1; i < 16; ++i) mm = fmaxf(mm, red[row * 16 + i]);
    red[row * 16] = mm;
  }
  __syncthreads();
  float mm = red[row * 16];
  float s = 0.f;
#pragma unroll
  for (int t = 0; t < 16; ++t) {
    int idx = row * 256 + sub + 16 * t;
    float e = __expf(sc[idx] - mm);
    sc[idx] = e;
    s += e;
  }
  __syncthreads();
  red[row * 16 + sub] = s;
  __syncthreads();
  if (sub == 0) {
    float ss = 0.f;
#pragma unroll
    for (int i = 0; i < 16; ++i) ss += red[row * 16 + i];
    red[row * 16] = ss;
  }
  __syncthreads();
  float inv = 1.0f / red[row * 16];
#pragma unroll
  for (int t = 0; t < 16; ++t) {
    int idx = row * 256 + sub + 16 * t;
    at[idx] = (bf16_t)(sc[idx] * inv);
  }

  // vp copy must be complete (own wave's chunks) before the LDS barrier
  wait_asynccnt0();
  __syncthreads();

  // --- z = attn @ vp : waves 0..3 take one 16-col E tile each, all from LDS ---
  if (wave < 4) {
    const bf16_t* VPl = vpl + (size_t)(wave * 16) * KK;  // (E,K) == BT, in LDS
    v8f acc = {};
#pragma unroll
    for (int kk = 0; kk < KK; kk += 32) {
      v16bf a = load_a_frag(at + kk, 256, lane);   // ds_load_b128
      v16bf b = load_b_frag(VPl + kk, KK, lane);   // ds_load_b128
      acc = wmma_bf16(a, b, acc);
    }
#pragma unroll
    for (int r = 0; r < 8; ++r) {
      int n = mt * 16 + r + ((lane >> 4) << 3);
      zb[(size_t)n * (HH * EE) + h * EE + wave * 16 + (lane & 15)] = (bf16_t)acc[r];
    }
  }
}

// out = z_concat @ Wo^T, f32 output (N, E). One wave per 16x64 strip.
__global__ void out_kernel(const bf16_t* __restrict__ zb,
                           const bf16_t* __restrict__ Wob,
                           float* __restrict__ out) {
  int lane = threadIdx.x & 31;
  int mt = blockIdx.x * (blockDim.x >> 5) + (threadIdx.x >> 5);
  if (mt >= NN / 16) return;

  const bf16_t* A0 = zb + (size_t)(mt * 16) * (HH * EE);
  const bf16_t* B0 = Wob;   // (E, H*E) row-major == BT

  v8f acc[4] = {};
  for (int kk = 0; kk < HH * EE; kk += 32) {
    __builtin_prefetch(A0 + kk + 256, 0, 3);
    __builtin_prefetch(B0 + kk + 256, 0, 3);
    v16bf a = load_a_frag(A0 + kk, HH * EE, lane);
#pragma unroll
    for (int e = 0; e < 4; ++e) {
      v16bf b = load_b_frag(B0 + (size_t)(e * 16) * (HH * EE) + kk, HH * EE, lane);
      acc[e] = wmma_bf16(a, b, acc[e]);
    }
  }
#pragma unroll
  for (int e = 0; e < 4; ++e) {
#pragma unroll
    for (int r = 0; r < 8; ++r) {
      int mrow = mt * 16 + r + ((lane >> 4) << 3);
      out[(size_t)mrow * EE + e * 16 + (lane & 15)] = acc[e][r];
    }
  }
}

// ---------------------------------------------------------------------------
// Host launcher
// ---------------------------------------------------------------------------
extern "C" void kernel_launch(void* const* d_in, const int* in_sizes, int n_in,
                              void* d_out, int out_size, void* d_ws, size_t ws_size,
                              hipStream_t stream) {
  (void)in_sizes; (void)n_in; (void)out_size; (void)ws_size;
  const float* x  = (const float*)d_in[0];
  const float* Wq = (const float*)d_in[1];
  const float* bq = (const float*)d_in[2];
  const float* Wk = (const float*)d_in[3];
  const float* bk = (const float*)d_in[4];
  const float* Wv = (const float*)d_in[5];
  const float* bv = (const float*)d_in[6];
  const float* Ew = (const float*)d_in[7];
  const float* Eb = (const float*)d_in[8];
  const float* Fw = (const float*)d_in[9];
  const float* Fb = (const float*)d_in[10];
  const float* Wo = (const float*)d_in[11];
  float* out = (float*)d_out;

  bf16_t* ws = (bf16_t*)d_ws;
  size_t off = 0;
  bf16_t* xb   = ws + off; off += (size_t)NN * DD;        // 4M
  bf16_t* Wqb  = ws + off; off += (size_t)HH * EE * DD;   // 1M
  bf16_t* Wkb  = ws + off; off += (size_t)HH * EE * DD;   // 1M
  bf16_t* Wvb  = ws + off; off += (size_t)HH * EE * DD;   // 1M
  bf16_t* Ewb  = ws + off; off += (size_t)HH * KK * NN;   // 16M
  bf16_t* Fwb  = ws + off; off += (size_t)HH * KK * NN;   // 16M
  bf16_t* Wob  = ws + off; off += (size_t)EE * HH * EE;   // 64K
  bf16_t* qb   = ws + off; off += (size_t)HH * NN * EE;   // 4M
  bf16_t* kTb  = ws + off; off += (size_t)HH * EE * NN;   // 4M
  bf16_t* vTb  = ws + off; off += (size_t)HH * EE * NN;   // 4M
  bf16_t* kpb  = ws + off; off += (size_t)HH * KK * EE;   // 256K
  bf16_t* vpTb = ws + off; off += (size_t)HH * EE * KK;   // 256K
  bf16_t* zb   = ws + off; off += (size_t)NN * HH * EE;   // 4M

  // fp32 -> bf16 casts (8 elems/thread)
  cast_f32_bf16_kernel<<<1024, 256, 0, stream>>>(x,  xb,  (size_t)NN * DD);
  cast_f32_bf16_kernel<<<256,  256, 0, stream>>>(Wq, Wqb, (size_t)HH * EE * DD);
  cast_f32_bf16_kernel<<<256,  256, 0, stream>>>(Wk, Wkb, (size_t)HH * EE * DD);
  cast_f32_bf16_kernel<<<256,  256, 0, stream>>>(Wv, Wvb, (size_t)HH * EE * DD);
  cast_f32_bf16_kernel<<<2048, 256, 0, stream>>>(Ew, Ewb, (size_t)HH * KK * NN);
  cast_f32_bf16_kernel<<<2048, 256, 0, stream>>>(Fw, Fwb, (size_t)HH * KK * NN);
  cast_f32_bf16_kernel<<<32,   256, 0, stream>>>(Wo, Wob, (size_t)EE * HH * EE);

  // q/k/v projections: 3*16*128 = 6144 wave-tiles (32x64 each), 8 waves/block
  qkv_kernel<<<768, 256, 0, stream>>>(xb, Wqb, Wkb, Wvb, bq, bk, bv, qb, kTb, vTb);

  // kp/vp: 2*16*16 = 512 wave-tiles (16x64 each)
  kpvp_kernel<<<64, 256, 0, stream>>>(Ewb, Fwb, kTb, vTb, Eb, Fb, kpb, vpTb);

  // fused score/softmax/z: one block per (head, 16-row strip)
  attn_kernel<<<HH * (NN / 16), 256, 0, stream>>>(qb, kpb, vpTb, zb);

  // output projection: 256 wave-tiles (16x64 each)
  out_kernel<<<32, 256, 0, stream>>>(zb, Wob, out);
}
